// Prior_19018115187058
// MI455X (gfx1250) — matrix operand order repaired and checked by reading, not compiled, not run on hardware
//
#include <hip/hip_runtime.h>

#define NPIX   131072      // B*H*W = 8*128*128
#define ZD     64
#define HD     256
#define MCODES 512
#define MUC    0.99f

typedef unsigned short u16;
typedef __attribute__((ext_vector_type(16))) __bf16         v16bf;
typedef __attribute__((ext_vector_type(8)))  unsigned int   v8u;
typedef __attribute__((ext_vector_type(8)))  float          v8f;
typedef __attribute__((ext_vector_type(4)))  unsigned int   u32x4;
typedef __attribute__((ext_vector_type(8)))  int            i32x8;
typedef __attribute__((ext_vector_type(4)))  int            i32x4;

__device__ __forceinline__ u16 f2bf(float f) {
  unsigned u = __float_as_uint(f);
  unsigned r = u + 0x7FFFu + ((u >> 16) & 1u);   // round-to-nearest-even
  return (u16)(r >> 16);
}
__device__ __forceinline__ float bf2f(unsigned h) {
  return __uint_as_float(h << 16);
}
__device__ __forceinline__ unsigned packbf2(float a, float b) {
  return (unsigned)f2bf(a) | ((unsigned)f2bf(b) << 16);
}

// ---------------------------------------------------------------------------
// WMMA fragment-order global layouts (bf16):
//  A-layout for [N rows, C cols]: chunk (nblk = n/16, kslab = c/32) of 512 u16
//    at ((nblk*(C/32)) + kslab)*512 ; within chunk: lane*16 + e, where for
//    kk = c%32:  lane = (n%16) + 16*((kk>>3)&1) ,  e = ((kk>>4)<<3) | (kk&7)
//  B-layout for [K rows, C cols]: chunk (kslab = k/32, nfrag = c/16) of 512 u16
//    at (kslab*(C/16) + nfrag)*512 ; within: lane = (c%16) + 16*((kk>>3)&1),
//    e = ((kk>>4)<<3) | (kk&7)   (kk = k%32)
// ---------------------------------------------------------------------------
__device__ __forceinline__ v16bf load_frag(const u16* p) {
  return __builtin_bit_cast(v16bf, *(const v8u*)p);
}

// ---------------------------------------------------------------------------
// Tensor Data Mover: 2D tile (rows x tile_qw0 qwords) global -> LDS.
// D# built per CDNA5 ISA §8 (data_size=8B, no pad/iterate/gather, type=2).
// Issue from one wave; completion via TENSORcnt.
// ---------------------------------------------------------------------------
__device__ __forceinline__ void tdm_load_2d(unsigned lds_off, const void* gaddr,
                                            unsigned tile_qw0,
                                            unsigned tile_rows,
                                            unsigned row_stride_qw) {
  const unsigned long long ga = (unsigned long long)gaddr;
  u32x4 g0;
  g0[0] = 1u;                                        // count=1, user D#
  g0[1] = lds_off;                                   // lds_addr (bytes)
  g0[2] = (unsigned)(ga & 0xFFFFFFFFu);              // global_addr[31:0]
  g0[3] = (unsigned)((ga >> 32) & 0x01FFFFFFu)       // global_addr[56:32]
          | (2u << 30);                              // type=2 ("image")
  i32x8 g1;
  g1[0] = (int)(3u << 16);                           // data_size=3 -> 8 bytes
  g1[1] = (int)((row_stride_qw & 0xFFFFu) << 16);    // tensor_dim0[15:0]
  g1[2] = (int)(((row_stride_qw >> 16) & 0xFFFFu)    // tensor_dim0[31:16]
                | ((tile_rows & 0xFFFFu) << 16));    // tensor_dim1[15:0]
  g1[3] = (int)(((tile_rows >> 16) & 0xFFFFu)        // tensor_dim1[31:16]
                | ((tile_qw0 & 0xFFFFu) << 16));     // tile_dim0
  g1[4] = (int)(tile_rows & 0xFFFFu);                // tile_dim1 (tile_dim2=0)
  g1[5] = (int)row_stride_qw;                        // tensor_dim0_stride lo32
  g1[6] = 0;                                         // stride hi / dim1_stride
  g1[7] = 0;
  const i32x4 gz4 = {0, 0, 0, 0};                    // <=2D: groups 2,3 unused
  const i32x8 gz8 = {0, 0, 0, 0, 0, 0, 0, 0};
  __builtin_amdgcn_tensor_load_to_lds(g0, g1, gz4, gz4, gz8, 0);
}
__device__ __forceinline__ unsigned lds_offset_of(const void* p) {
  return (unsigned)(unsigned long long)p;            // flat LDS: low 32 bits
}

// ---------------- weight prep: w[cout,cin] f32 -> B-fragment-layout bf16 ----
__global__ void __launch_bounds__(256)
wprep_kernel(const float* __restrict__ w, u16* __restrict__ wt,
             int cout, int cin) {
  int idx = blockIdx.x * 256 + threadIdx.x;
  if (idx < cout * cin) {
    const int o = idx / cin, k = idx % cin;           // col = o, K-row = k
    const int ks = k >> 5, kk = k & 31;
    const int nf = o >> 4;
    const int lane = (o & 15) + 16 * ((kk >> 3) & 1);
    const int e = ((kk >> 4) << 3) | (kk & 7);
    wt[(((size_t)ks * (cout >> 4) + nf) << 9) + lane * 16 + e] = f2bf(w[idx]);
  }
}

// ---------------- centroid prep --------------------------------------------
__global__ void __launch_bounds__(64)
centroid_kernel(const float* __restrict__ ps, const float* __restrict__ pe,
                float* __restrict__ out_e, u16* __restrict__ etf,
                float* __restrict__ enorm, float* __restrict__ sum_acc,
                float* __restrict__ cnt_acc) {
  __shared__ float red[64];
  const int m = blockIdx.x, c = threadIdx.x;          // code m, dim c
  const float ev = ps[m * ZD + c] / pe[m];
  out_e[m * ZD + c] = ev;
  {                                                   // e^T in B-fragment order
    const int ks = c >> 5, kk = c & 31;
    const int nf = m >> 4;
    const int lane = (m & 15) + 16 * ((kk >> 3) & 1);
    const int e = ((kk >> 4) << 3) | (kk & 7);
    etf[(((size_t)ks * (MCODES >> 4) + nf) << 9) + lane * 16 + e] = f2bf(ev);
  }
  sum_acc[m * ZD + c] = 0.0f;                         // zero EMA accum each call
  if (c == 0) cnt_acc[m] = 0.0f;
  red[c] = ev * ev;
  __syncthreads();
#pragma unroll
  for (int s = 32; s >= 1; s >>= 1) {
    if (c < s) red[c] += red[c + s];
    __syncthreads();
  }
  if (c == 0) enorm[m] = red[0];
}

// ---------------- z NCHW -> A-fragment-layout bf16 --------------------------
__global__ void __launch_bounds__(256)
zprep_kernel(const float* __restrict__ z, u16* __restrict__ zt) {
  __shared__ u16 tile[64][65];                        // [channel][pixel]
  const int n0 = blockIdx.x * 64;
  const int b = n0 >> 14, h = (n0 >> 7) & 127, w0 = n0 & 127;
  const int t = threadIdx.x;
#pragma unroll
  for (int i = 0; i < 16; ++i) {
    const int idx = i * 256 + t;
    const int c = idx >> 6, wd = idx & 63;            // coalesced along w
    tile[c][wd] = f2bf(z[(((size_t)b * 64 + c) * 128 + h) * 128 + w0 + wd]);
  }
  __syncthreads();
  // write A-fragment layout: 8 chunks (rb 0..3 x ks2 0..1), 256 u32 each
#pragma unroll
  for (int i = 0; i < 8; ++i) {
    const int idx = i * 256 + t;                      // u32 index 0..2047
    const int chunk = idx >> 8, within = idx & 255;
    const int rb = chunk >> 1, ks2 = chunk & 1;
    const int lane = within >> 3, ep = (within & 7) * 2;
    const int r = rb * 16 + (lane & 15);
    const int half = lane >> 4;
    const int kk = (ep & 7) | (half << 3) | ((ep >> 3) << 4);
    const int c = ks2 * 32 + kk;
    const unsigned pk = (unsigned)tile[c][r] | ((unsigned)tile[c + 1][r] << 16);
    *(unsigned*)&zt[((((size_t)(n0 >> 4) + rb) * (ZD >> 5) + ks2) << 9) +
                    within * 2] = pk;
  }
}

// ---------------- WMMA GEMM: O = act(A[N,CIN] * W[CIN,COUT] + bias) ---------
template <int CIN, int COUT, bool FINAL>
__global__ void __launch_bounds__(128)
gemm_kernel(const u16* __restrict__ A, const u16* __restrict__ W,
            const float* __restrict__ bias, u16* __restrict__ O,
            float* __restrict__ zq) {
  constexpr int KS = CIN / 32;
  __shared__ u16 Bs[KS * 4 * 512];                    // B panel (<=32KB)
  __shared__ float Stg[64 * 64];

  const int n0 = blockIdx.x * 64;
  const int c0 = blockIdx.y * 64;
  const int t = threadIdx.x;
  const int lane = t & 31;
  const int wv = t >> 5;

  // stage B panel via Tensor Data Mover: KS rows x 4KB, row stride = panel row
  if (wv == 0) {
    tdm_load_2d(lds_offset_of(&Bs[0]), W + ((size_t)(c0 >> 4) << 9),
                /*tile_qw0=*/512, /*tile_rows=*/KS,
                /*row_stride_qw=*/COUT * 8);
    __builtin_amdgcn_s_wait_tensorcnt(0);
  }
  __syncthreads();

  v8f acc[4];
#pragma unroll
  for (int f = 0; f < 4; ++f)
#pragma unroll
    for (int i = 0; i < 8; ++i) acc[f][i] = 0.0f;

  const u16* aBase =
      A + (((size_t)(n0 >> 4) + wv) * KS << 9) + lane * 16;
#pragma unroll
  for (int ks = 0; ks < KS; ++ks) {
    if (ks + 1 < KS) __builtin_prefetch(aBase + ((ks + 1) << 9), 0, 0);
    const v16bf a = load_frag(aBase + (ks << 9));
#pragma unroll
    for (int f = 0; f < 4; ++f) {
      const v16bf b = load_frag(&Bs[(ks * 4 + f) * 512 + lane * 16]);
      acc[f] = __builtin_amdgcn_wmma_f32_16x16x32_bf16(
          false, a, false, b, (short)0, acc[f], false, false);
    }
  }

  // epilogue: bias (+tanh) -> f32 LDS staging tile
  const int colLane = lane & 15;
  const int rhalf = (lane >> 4) * 8;
#pragma unroll
  for (int f = 0; f < 4; ++f) {
    const int col = f * 16 + colLane;
    const float bv = bias[c0 + col];
#pragma unroll
    for (int v = 0; v < 8; ++v) {
      float x = acc[f][v] + bv;
      if (!FINAL) x = tanhf(x);
      Stg[(wv * 16 + rhalf + v) * 64 + col] = x;
    }
  }
  __syncthreads();

  // store output in A-fragment layout (packed bf16, coalesced u32)
#pragma unroll
  for (int i = 0; i < 16; ++i) {
    const int idx = i * 128 + t;                      // u32 index 0..2047
    const int chunk = idx >> 8, within = idx & 255;
    const int rb = chunk >> 1, ks2 = chunk & 1;
    const int lane2 = within >> 3, ep = (within & 7) * 2;
    const int r = rb * 16 + (lane2 & 15);
    const int half = lane2 >> 4;
    const int kk = (ep & 7) | (half << 3) | ((ep >> 3) << 4);
    const int c = ks2 * 32 + kk;
    const unsigned pk = packbf2(Stg[r * 64 + c], Stg[r * 64 + c + 1]);
    *(unsigned*)&O[((((size_t)(n0 >> 4) + rb) * (COUT >> 5) + (c0 >> 5) + ks2)
                    << 9) +
                   within * 2] = pk;
  }

  if (FINAL) {                                        // zq [B,C,H,W] f32
    const int b = n0 >> 14, h = (n0 >> 7) & 127, w0 = n0 & 127;
#pragma unroll
    for (int i = 0; i < 8; ++i) {
      const int idx = i * 128 + t;                    // 1024 float4
      const int c = idx >> 4, p4 = idx & 15;
      float4 o;
      o.x = Stg[(p4 * 4 + 0) * 64 + c];
      o.y = Stg[(p4 * 4 + 1) * 64 + c];
      o.z = Stg[(p4 * 4 + 2) * 64 + c];
      o.w = Stg[(p4 * 4 + 3) * 64 + c];
      *(float4*)&zq[(((size_t)(b * 64 + c) * 128) + h) * 128 + w0 + p4 * 4] = o;
    }
  }
}

// ---------------- distance + argmin + one-hot + EMA stats -------------------
__global__ void __launch_bounds__(128)
dist_kernel(const u16* __restrict__ zef, const u16* __restrict__ etf,
            const float* __restrict__ enorm, float* __restrict__ onehot,
            float* __restrict__ sum_acc, float* __restrict__ cnt_acc) {
  __shared__ u16 Bs[2 * 4 * 512];                     // 8KB e^T chunk
  __shared__ int zi_s[64];

  const int n0 = blockIdx.x * 64;
  const int t = threadIdx.x, lane = t & 31, wv = t >> 5;

  const u16* aBase =
      zef + (((size_t)(n0 >> 4) + wv) * 2 << 9) + lane * 16;
  const v16bf a0 = load_frag(aBase);
  const v16bf a1 = load_frag(aBase + 512);

  float best[8];
  int bidx[8];
#pragma unroll
  for (int v = 0; v < 8; ++v) { best[v] = 3.4e38f; bidx[v] = 0; }

  for (int cc = 0; cc < MCODES / 64; ++cc) {
    const int c0 = cc * 64;
    __syncthreads();                                  // previous chunk consumed
    if (wv == 0) {                                    // TDM: 2 rows x 4KB
      tdm_load_2d(lds_offset_of(&Bs[0]), etf + ((size_t)(c0 >> 4) << 9),
                  /*tile_qw0=*/512, /*tile_rows=*/2,
                  /*row_stride_qw=*/MCODES * 8);
      __builtin_amdgcn_s_wait_tensorcnt(0);
    }
    __syncthreads();

    v8f acc[4];
#pragma unroll
    for (int f = 0; f < 4; ++f)
#pragma unroll
      for (int i = 0; i < 8; ++i) acc[f][i] = 0.0f;
#pragma unroll
    for (int ks = 0; ks < 2; ++ks) {
      const v16bf a = ks ? a1 : a0;
#pragma unroll
      for (int f = 0; f < 4; ++f) {
        const v16bf b = load_frag(&Bs[(ks * 4 + f) * 512 + lane * 16]);
        acc[f] = __builtin_amdgcn_wmma_f32_16x16x32_bf16(
            false, a, false, b, (short)0, acc[f], false, false);
      }
    }
    // dist_eff = ||e||^2 - 2*dot   (||ze||^2 constant per row, dropped)
#pragma unroll
    for (int f = 0; f < 4; ++f) {
      const int col = c0 + f * 16 + (lane & 15);
      const float en = enorm[col];
#pragma unroll
      for (int v = 0; v < 8; ++v) {
        const float d = __builtin_fmaf(-2.0f, acc[f][v], en);
        if (d < best[v]) { best[v] = d; bidx[v] = col; }
      }
    }
  }

  // cross-lane argmin within each 16-lane half (lexicographic tie-break)
#pragma unroll
  for (int v = 0; v < 8; ++v) {
    float bv = best[v];
    int bi = bidx[v];
#pragma unroll
    for (int mask = 8; mask >= 1; mask >>= 1) {
      const float ov = __shfl_xor(bv, mask, 16);
      const int oi = __shfl_xor(bi, mask, 16);
      if (ov < bv || (ov == bv && oi < bi)) { bv = ov; bi = oi; }
    }
    if ((lane & 15) == 0) zi_s[wv * 16 + (lane >> 4) * 8 + v] = bi;
  }
  __syncthreads();

  // one-hot rows: 64 rows x 512 f32, one float4 per thread per row
  for (int r = 0; r < 64; ++r) {
    const int code = zi_s[r];
    const int cb = t * 4;
    float4 o;
    o.x = (cb + 0 == code) ? 1.0f : 0.0f;
    o.y = (cb + 1 == code) ? 1.0f : 0.0f;
    o.z = (cb + 2 == code) ? 1.0f : 0.0f;
    o.w = (cb + 3 == code) ? 1.0f : 0.0f;
    *(float4*)&onehot[(size_t)(n0 + r) * MCODES + cb] = o;
  }

  // EMA stats via coalesced fragment-order walk of ze (paired bf16)
#pragma unroll
  for (int i = 0; i < 16; ++i) {
    const int idx = i * 128 + t;                      // u32 index 0..2047
    const int chunk = idx >> 8, within = idx & 255;
    const int rb = chunk >> 1, ks2 = chunk & 1;
    const int lane2 = within >> 3, ep = (within & 7) * 2;
    const int r = rb * 16 + (lane2 & 15);
    const int half = lane2 >> 4;
    const int kk = (ep & 7) | (half << 3) | ((ep >> 3) << 4);
    const int c = ks2 * 32 + kk;
    const unsigned pk =
        *(const unsigned*)&zef[((((size_t)(n0 >> 4) + rb) * 2 + ks2) << 9) +
                               within * 2];
    const int code = zi_s[r];
    atomicAdd(&sum_acc[(size_t)code * ZD + c], bf2f(pk & 0xFFFFu));
    atomicAdd(&sum_acc[(size_t)code * ZD + c + 1], bf2f(pk >> 16));
  }
  if (t < 64) atomicAdd(&cnt_acc[zi_s[t]], 1.0f);
}

// ---------------- EMA finalize ---------------------------------------------
__global__ void __launch_bounds__(256)
final_kernel(const float* __restrict__ ps, const float* __restrict__ pe,
             const float* __restrict__ sa, const float* __restrict__ ca,
             float* __restrict__ ops, float* __restrict__ ope) {
  const int idx = blockIdx.x * 256 + threadIdx.x;
  if (idx < MCODES * ZD) ops[idx] = MUC * ps[idx] + (1.0f - MUC) * sa[idx];
  if (idx < MCODES) ope[idx] = MUC * pe[idx] + (1.0f - MUC) * ca[idx];
}

// ---------------- host launch ----------------------------------------------
extern "C" void kernel_launch(void* const* d_in, const int* in_sizes, int n_in,
                              void* d_out, int out_size, void* d_ws,
                              size_t ws_size, hipStream_t stream) {
  (void)in_sizes; (void)n_in; (void)out_size; (void)ws_size;
  const float* z  = (const float*)d_in[0];
  const float* ps = (const float*)d_in[1];
  const float* pe = (const float*)d_in[2];
  const float* w1 = (const float*)d_in[3];
  const float* b1 = (const float*)d_in[4];
  const float* w2 = (const float*)d_in[5];
  const float* b2 = (const float*)d_in[6];
  const float* w3 = (const float*)d_in[7];
  const float* b3 = (const float*)d_in[8];
  const float* w4 = (const float*)d_in[9];
  const float* b4 = (const float*)d_in[10];

  float* out    = (float*)d_out;
  float* out_e  = out;                       // [512,64]
  float* out_zq = out + 32768;               // [8,64,128,128]
  float* out_oh = out + 557056;              // [131072,512]
  float* out_ps = out + 67665920;            // [512,64]
  float* out_pe = out + 67698688;            // [512]

  char* wsb = (char*)d_ws;
  size_t off = 0;
  auto alloc = [&](size_t bytes) -> void* {
    void* p = wsb + off;
    off += (bytes + 255) & ~(size_t)255;
    return p;
  };
  u16* zt   = (u16*)alloc((size_t)NPIX * ZD * 2);   // reused later as ze
  u16* hA   = (u16*)alloc((size_t)NPIX * HD * 2);
  u16* hB   = (u16*)alloc((size_t)NPIX * HD * 2);
  u16* w1t  = (u16*)alloc((size_t)ZD * HD * 2);
  u16* w2t  = (u16*)alloc((size_t)HD * HD * 2);
  u16* w3t  = (u16*)alloc((size_t)HD * HD * 2);
  u16* w4t  = (u16*)alloc((size_t)HD * ZD * 2);
  u16* etf  = (u16*)alloc((size_t)ZD * MCODES * 2);
  float* enorm   = (float*)alloc(MCODES * 4);
  float* sum_acc = (float*)alloc((size_t)MCODES * ZD * 4);
  float* cnt_acc = (float*)alloc(MCODES * 4);
  u16* zeb = zt;                                    // zt consumed by gemm1

  wprep_kernel<<<(HD * ZD + 255) / 256, 256, 0, stream>>>(w1, w1t, HD, ZD);
  wprep_kernel<<<(HD * HD + 255) / 256, 256, 0, stream>>>(w2, w2t, HD, HD);
  wprep_kernel<<<(HD * HD + 255) / 256, 256, 0, stream>>>(w3, w3t, HD, HD);
  wprep_kernel<<<(ZD * HD + 255) / 256, 256, 0, stream>>>(w4, w4t, ZD, HD);
  centroid_kernel<<<MCODES, 64, 0, stream>>>(ps, pe, out_e, etf, enorm,
                                             sum_acc, cnt_acc);
  zprep_kernel<<<NPIX / 64, 256, 0, stream>>>(z, zt);

  gemm_kernel<ZD, HD, false><<<dim3(NPIX / 64, HD / 64), 128, 0, stream>>>(
      zt, w1t, b1, hA, nullptr);
  gemm_kernel<HD, HD, false><<<dim3(NPIX / 64, HD / 64), 128, 0, stream>>>(
      hA, w2t, b2, hB, nullptr);
  gemm_kernel<HD, HD, false><<<dim3(NPIX / 64, HD / 64), 128, 0, stream>>>(
      hB, w3t, b3, hA, nullptr);
  gemm_kernel<HD, ZD, true><<<dim3(NPIX / 64, 1), 128, 0, stream>>>(
      hA, w4t, b4, zeb, out_zq);

  dist_kernel<<<NPIX / 64, 128, 0, stream>>>(zeb, etf, enorm, out_oh,
                                             sum_acc, cnt_acc);
  final_kernel<<<(MCODES * ZD + 255) / 256, 256, 0, stream>>>(
      ps, pe, sum_acc, cnt_acc, out_ps, out_pe);
}
